// TrustGatesSTLSTMModel_26800595927215
// MI455X (gfx1250) — compile-verified, matrix operand order checked by reading, not run on hardware
//
#include <hip/hip_runtime.h>
#include <hip/hip_bf16.h>
#include <math.h>

// ---------------------------------------------------------------------------
// ST-LSTM wavefront implementation for gfx1250 (MI455X).
//   Grid of cells (s,t), s<25, t<100. Diagonal phases p = s+t are independent.
//   Recurrent GEMM (256x640, K=256) via v_wmma_f32_16x16x32_bf16.
//   State H stored bf16 (A-operand ready), C stored f32. Double buffered by
//   phase parity with an extra zero slot for the s=0 / t=0 boundaries.
// ---------------------------------------------------------------------------

typedef __bf16 bf16;
typedef __attribute__((ext_vector_type(16))) __bf16 v16bf;
typedef __attribute__((ext_vector_type(8)))  __bf16 v8bf;
typedef __attribute__((ext_vector_type(8)))  float  v8f;

#define BQ 256      // batch
#define TT 100      // time steps
#define SS 25       // spatial steps
#define DD 3        // input dim
#define HH 128      // hidden
#define NG 640      // 5*H
#define CC 60       // classes
#define MT 16       // rows per workgroup

#define SLOT (BQ*HH)                 // elements per (s)-slot of state
#define HALF_SLOTS (SS+1)            // +1 zero slot for boundary
// workspace byte offsets (all multiples of 256)
#define OFF_HARR   0u
#define SZ_HARR    (2u*HALF_SLOTS*SLOT*2u)          // bf16, 2 parities = 3,407,872
#define OFF_CARR   (OFF_HARR + SZ_HARR)
#define SZ_CARR    (2u*HALF_SLOTS*SLOT*4u)          // f32             = 6,815,744
#define OFF_LOGP   (OFF_CARR + SZ_CARR)
#define SZ_LOGP    (SS*BQ*CC*4u)                    //                 = 1,536,000
#define OFF_LOSS   (OFF_LOGP + SZ_LOGP)
#define SZ_LOSS    2048u                            // 25*16 floats, padded
#define OFF_WPACK  (OFF_LOSS + SZ_LOSS)
#define SZ_WPACK   (8u*40u*32u*16u*2u)              // packed bf16 B-tiles = 327,680

// ---------------------------------------------------------------------------
// Zero-init the accumulating / state regions of the workspace.
__global__ __launch_bounds__(256) void zero_ws(float* ws, int ndw) {
    int i = blockIdx.x * 256 + threadIdx.x;
    int stride = gridDim.x * 256;
    for (; i < ndw; i += stride) ws[i] = 0.0f;
}

// ---------------------------------------------------------------------------
// Pack [Whs;Wht] (256x640 f32, row-major) into per-lane WMMA B-tile layout:
//   Wpack[kt][nt][lane][e] (bf16), kt in [0,8) (K=32 each), nt in [0,40).
//   B 32x16 bf16 layout: lane holds column n = nt*16 + (lane&15); lane-half
//   selects k-subblocks: e<8 -> k = 8*half + e ; e>=8 -> k = 16 + 8*half + (e-8).
__global__ __launch_bounds__(256) void pack_weights(const float* __restrict__ Whs,
                                                    const float* __restrict__ Wht,
                                                    bf16* __restrict__ Wpack) {
    int idx = blockIdx.x * 256 + threadIdx.x;
    if (idx >= 8 * 40 * 32 * 16) return;
    int e    = idx & 15;
    int lane = (idx >> 4) & 31;
    int rest = idx >> 9;
    int nt   = rest % 40;
    int kt   = rest / 40;
    int lh   = lane >> 4;
    int n    = nt * 16 + (lane & 15);
    int kloc = (e & 7) + 8 * lh + 16 * (e >> 3);   // 0..31
    int kg   = kt * 32 + kloc;                     // 0..255 over [hs|ht]
    float w  = (kg < HH) ? Whs[kg * NG + n] : Wht[(kg - HH) * NG + n];
    Wpack[idx] = (bf16)w;
}

// ---------------------------------------------------------------------------
// One diagonal phase: all cells (s, t=p-s). blockIdx.x = di*16 + mtile.
__global__ __launch_bounds__(256) void cell_phase(
    const float* __restrict__ x,   const int* __restrict__ y,
    const float* __restrict__ Wx,  const float* __restrict__ bvec,
    const float* __restrict__ Wfc, const float* __restrict__ bfc,
    const bf16* __restrict__ Wpack,
    bf16* __restrict__ Harr, float* __restrict__ Carr,
    float* __restrict__ logpacc, float* __restrict__ lossacc,
    int p, int sLo)
{
    const int di = blockIdx.x >> 4;
    const int mtile = blockIdx.x & 15;
    const int s = sLo + di;
    const int t = p - s;
    if (s >= SS || t < 0 || t >= TT) return;   // uniform per block
    const int m0 = mtile * MT;

    const int wp = p & 1, rp = wp ^ 1;
    const bf16*  Hr = Harr + (size_t)rp * HALF_SLOTS * SLOT;
    bf16*        Hw = Harr + (size_t)wp * HALF_SLOTS * SLOT;
    const float* Cr = Carr + (size_t)rp * HALF_SLOTS * SLOT;
    float*       Cw = Carr + (size_t)wp * HALF_SLOTS * SLOT;

    __shared__ float smem[MT * NG + MT * HH + MT * DD + MT];
    float* zS = smem;                 // 16x640 z / reused as 16x60 logits
    float* hS = smem + MT * NG;       // 16x128 h (f32)
    float* xS = hS + MT * HH;         // 16x3 x tile
    float* rowloss = xS + MT * DD;    // 16 per-row loss terms

    // stage x tile (16 rows x 3)
    for (int i = threadIdx.x; i < MT * DD; i += 256) {
        int r = i / DD, d = i - r * DD;
        xS[i] = x[(size_t)(m0 + r) * (TT * SS * DD) + t * (SS * DD) + s * DD + d];
    }
    __syncthreads();

    const int wv = threadIdx.x >> 5;
    const int lane = threadIdx.x & 31;
    const int lh = lane >> 4;
    const int n0w = wv * 80;           // 5 N-tiles of 16 per wave
    const int arow = m0 + (lane & 15);

    v8f acc[5];
    #pragma unroll
    for (int j = 0; j < 5; ++j) acc[j] = (v8f)(0.0f);

    // K loop over 8 blocks of 32: kt<4 -> hs (slot s), kt>=4 -> ht (slot s+1)
    #pragma unroll
    for (int kt = 0; kt < 8; ++kt) {
        const int slot = s + (kt >> 2);
        const int kk = (kt & 3) * 32;
        const bf16* ap = Hr + (size_t)slot * SLOT + arow * HH + kk + 8 * lh;
        v8bf alo = *(const v8bf*)(ap);
        v8bf ahi = *(const v8bf*)(ap + 16);
        v16bf a = __builtin_shufflevector(alo, ahi,
                    0,1,2,3,4,5,6,7,8,9,10,11,12,13,14,15);
        const bf16* bp = Wpack + ((((size_t)kt * 40 + (n0w >> 4)) * 32 + lane) << 4);
        if (kt < 7)
            __builtin_prefetch(bp + (size_t)40 * 32 * 16, 0, 1);
        #pragma unroll
        for (int j = 0; j < 5; ++j) {
            v16bf bm = *(const v16bf*)(bp + (size_t)j * 32 * 16);
            acc[j] = __builtin_amdgcn_wmma_f32_16x16x32_bf16(
                false, a, false, bm, (short)0, acc[j], false, false);
        }
    }

    // epilogue: + bias + xt@Wx (D=3) -> zS
    #pragma unroll
    for (int j = 0; j < 5; ++j) {
        const int col = n0w + j * 16 + (lane & 15);
        const float wx0 = Wx[col], wx1 = Wx[NG + col], wx2 = Wx[2 * NG + col];
        const float bb = bvec[col];
        #pragma unroll
        for (int r = 0; r < 8; ++r) {
            const int M = r + 8 * lh;
            zS[M * NG + col] = acc[j][r] + bb
                + xS[M * DD + 0] * wx0 + xS[M * DD + 1] * wx1 + xS[M * DD + 2] * wx2;
        }
    }
    __syncthreads();

    // gates + state update: 16x128 elements
    for (int q = threadIdx.x; q < MT * HH; q += 256) {
        const int rI = q >> 7, cj = q & 127;
        const float* zr = zS + rI * NG;
        const float zi = zr[cj], zfs = zr[HH + cj], zft = zr[2 * HH + cj];
        const float zo = zr[3 * HH + cj], zg = zr[4 * HH + cj];
        const float gi = 1.0f / (1.0f + __expf(-zi));
        const float gfs = 1.0f / (1.0f + __expf(-zfs));
        const float gft = 1.0f / (1.0f + __expf(-zft));
        const float go = 1.0f / (1.0f + __expf(-zo));
        const float gg = tanhf(zg);
        const int row = m0 + rI;
        const float cs = Cr[(size_t)s * SLOT + row * HH + cj];          // C(s-1,t)
        const float ct = Cr[(size_t)(s + 1) * SLOT + row * HH + cj];    // C(s,t-1)
        const float c = gi * gg + gfs * cs + gft * ct;
        const float h = go * tanhf(c);
        Cw[(size_t)(s + 1) * SLOT + row * HH + cj] = c;
        Hw[(size_t)(s + 1) * SLOT + row * HH + cj] = (bf16)h;
        hS[rI * HH + cj] = h;
    }
    __syncthreads();

    // FC head 128->60 in f32 (zS reused as logits)
    float* lg = zS;
    for (int q = threadIdx.x; q < MT * CC; q += 256) {
        const int rI = q / CC, c = q - rI * CC;
        float a = bfc[c];
        const float* hp = hS + rI * HH;
        #pragma unroll 4
        for (int k = 0; k < HH; ++k) a += hp[k] * Wfc[k * CC + c];
        lg[rI * CC + c] = a;
    }
    __syncthreads();

    // log-softmax per row + accumulation
    if (threadIdx.x < MT) {
        const int rI = threadIdx.x;
        const float* lr = lg + rI * CC;
        float mx = -INFINITY;
        for (int c = 0; c < CC; ++c) mx = fmaxf(mx, lr[c]);
        float sum = 0.0f;
        for (int c = 0; c < CC; ++c) sum += __expf(lr[c] - mx);
        const float lse = mx + __logf(sum);
        float* op = logpacc + ((size_t)s * BQ + m0 + rI) * CC;
        for (int c = 0; c < CC; ++c) op[c] += lr[c] - lse;
        const int yb = y[m0 + rI];
        rowloss[rI] = lr[yb] - lse;
    }
    __syncthreads();
    if (threadIdx.x == 0) {
        float sl = 0.0f;
        for (int i = 0; i < MT; ++i) sl += rowloss[i];
        lossacc[s * 16 + mtile] += sl;   // unique (s,mtile) writer per phase
    }
}

// ---------------------------------------------------------------------------
__global__ __launch_bounds__(256) void finalize(const float* __restrict__ logpacc,
                                                const float* __restrict__ lossacc,
                                                float* __restrict__ out) {
    int idx = blockIdx.x * 256 + threadIdx.x;
    if (idx < BQ * CC) {
        float sum = 0.0f;
        for (int s = 0; s < SS; ++s) sum += logpacc[(size_t)s * BQ * CC + idx];
        out[idx] = sum * (1.0f / (SS * TT));
    } else if (idx == BQ * CC) {
        float sum = 0.0f;
        for (int i = 0; i < SS * 16; ++i) sum += lossacc[i];
        out[BQ * CC] = -sum / ((float)SS * TT * BQ);
    }
}

// ---------------------------------------------------------------------------
extern "C" void kernel_launch(void* const* d_in, const int* in_sizes, int n_in,
                              void* d_out, int out_size, void* d_ws, size_t ws_size,
                              hipStream_t stream) {
    const float* x   = (const float*)d_in[0];
    const int*   y   = (const int*)d_in[1];
    const float* Wx  = (const float*)d_in[2];
    const float* Whs = (const float*)d_in[3];
    const float* Wht = (const float*)d_in[4];
    const float* b   = (const float*)d_in[5];
    const float* Wfc = (const float*)d_in[6];
    const float* bfc = (const float*)d_in[7];

    char* ws = (char*)d_ws;
    bf16*  Harr    = (bf16*)(ws + OFF_HARR);
    float* Carr    = (float*)(ws + OFF_CARR);
    float* logpacc = (float*)(ws + OFF_LOGP);
    float* lossacc = (float*)(ws + OFF_LOSS);
    bf16*  Wpack   = (bf16*)(ws + OFF_WPACK);

    // zero state + accumulators (everything before Wpack)
    const int ndw = (int)(OFF_WPACK / 4);
    zero_ws<<<512, 256, 0, stream>>>((float*)ws, ndw);

    pack_weights<<<(8 * 40 * 32 * 16 + 255) / 256, 256, 0, stream>>>(Whs, Wht, Wpack);

    // wavefront over diagonals p = s + t
    for (int p = 0; p < SS + TT - 1; ++p) {
        const int sLo = (p - (TT - 1) > 0) ? (p - (TT - 1)) : 0;
        const int sHi = (p < SS - 1) ? p : (SS - 1);
        const int nd = sHi - sLo + 1;
        cell_phase<<<dim3(nd * 16), 256, 0, stream>>>(
            x, y, Wx, b, Wfc, bfc, Wpack, Harr, Carr, logpacc, lossacc, p, sLo);
    }

    finalize<<<(BQ * CC + 1 + 255) / 256, 256, 0, stream>>>(logpacc, lossacc, (float*)d_out);
}
